// SAModule_66168266162351
// MI455X (gfx1250) — matrix operand order repaired and checked by reading, not compiled, not run on hardware
//
#include <hip/hip_runtime.h>

#define B_    8
#define N_    4096
#define M_    1024
#define KN    64      // max neighbors
#define FIN   64
#define OUT_  128
#define R2_   0.04f   // R = 0.2
#define KPAD  96      // 67 padded to 3 k-steps of 32
#define FPS_T 1024    // threads per FPS workgroup (32 waves)

typedef __attribute__((ext_vector_type(16))) _Float16 v16h;
typedef __attribute__((ext_vector_type(8)))  float    v8f;

// ---------------------------------------------------------------------------
// Kernel 0: transpose + f16-convert weights into workspace (K-major per col).
// W1t[n][k] n<64,k<96(pad) ; W2t[n][k] n<64,k<64 ; W3t[n][k] n<128,k<64
// ---------------------------------------------------------------------------
__global__ void wconv_kernel(const float* __restrict__ W1,
                             const float* __restrict__ W2,
                             const float* __restrict__ W3,
                             _Float16* __restrict__ W1t,
                             _Float16* __restrict__ W2t,
                             _Float16* __restrict__ W3t) {
    int tid = threadIdx.x;
    for (int i = tid; i < 64 * KPAD; i += 256) {
        int n = i / KPAD, k = i % KPAD;
        W1t[i] = (k < 67) ? (_Float16)W1[k * 64 + n] : (_Float16)0.f;
    }
    for (int i = tid; i < 64 * 64; i += 256) {
        int n = i >> 6, k = i & 63;
        W2t[i] = (_Float16)W2[k * 64 + n];
    }
    for (int i = tid; i < 128 * 64; i += 256) {
        int n = i >> 6, k = i & 63;
        W3t[i] = (_Float16)W3[k * 128 + n];
    }
}

// ---------------------------------------------------------------------------
// Kernel 1: farthest point sampling, one workgroup per cloud.
// Positions cached in LDS (SoA); per-round argmax = wave-shuffle reduction
// (no barriers) + single 32-partial cross-wave pass -> 3 barriers/round.
// ---------------------------------------------------------------------------
__global__ void __launch_bounds__(FPS_T)
fps_kernel(const float* __restrict__ pos,
           unsigned short* __restrict__ sel,
           float* __restrict__ selpos,
           int* __restrict__ selbatch) {
    __shared__ float px[N_], py[N_], pz[N_];
    __shared__ float dmin[N_];
    __shared__ float wrv[FPS_T / 32];
    __shared__ int   wri[FPS_T / 32];
    __shared__ int   curs;

    int b = blockIdx.x, tid = threadIdx.x;
    const float* pb = pos + (size_t)b * N_ * 3;

    for (int j = tid; j < N_; j += FPS_T) {
        px[j] = pb[j * 3 + 0];
        py[j] = pb[j * 3 + 1];
        pz[j] = pb[j * 3 + 2];
    }
    for (int q = tid; q < M_; q += FPS_T) selbatch[b * M_ + q] = b;
    __syncthreads();

    float p0x = px[0], p0y = py[0], p0z = pz[0];
    for (int j = tid; j < N_; j += FPS_T) {
        float dx = px[j] - p0x, dy = py[j] - p0y, dz = pz[j] - p0z;
        dmin[j] = dx * dx + dy * dy + dz * dz;
    }
    if (tid == 0) {
        sel[b * M_] = 0;
        selpos[(size_t)(b * M_) * 3 + 0] = p0x;
        selpos[(size_t)(b * M_) * 3 + 1] = p0y;
        selpos[(size_t)(b * M_) * 3 + 2] = p0z;
    }
    __syncthreads();

    for (int t = 1; t < M_; ++t) {
        // local scan: 4 elements per thread
        float bv = -1.f; int bi = 0;
        for (int j = tid; j < N_; j += FPS_T)
            if (dmin[j] > bv) { bv = dmin[j]; bi = j; }
        // wave32 shuffle argmax (tie -> lower index)
        for (int off = 16; off > 0; off >>= 1) {
            float ov = __shfl_down(bv, off);
            int   oi = __shfl_down(bi, off);
            if (ov > bv || (ov == bv && oi < bi)) { bv = ov; bi = oi; }
        }
        if ((tid & 31) == 0) { wrv[tid >> 5] = bv; wri[tid >> 5] = bi; }
        __syncthreads();
        if (tid < 32) {
            bv = wrv[tid]; bi = wri[tid];
            for (int off = 16; off > 0; off >>= 1) {
                float ov = __shfl_down(bv, off);
                int   oi = __shfl_down(bi, off);
                if (ov > bv || (ov == bv && oi < bi)) { bv = ov; bi = oi; }
            }
            if (tid == 0) {
                curs = bi;
                sel[b * M_ + t] = (unsigned short)bi;
                selpos[(size_t)(b * M_ + t) * 3 + 0] = px[bi];
                selpos[(size_t)(b * M_ + t) * 3 + 1] = py[bi];
                selpos[(size_t)(b * M_ + t) * 3 + 2] = pz[bi];
            }
        }
        __syncthreads();
        int ci = curs;
        float cx = px[ci], cy = py[ci], cz = pz[ci];
        for (int j = tid; j < N_; j += FPS_T) {
            float dx = px[j] - cx, dy = py[j] - cy, dz = pz[j] - cz;
            float d2 = dx * dx + dy * dy + dz * dz;
            if (d2 < dmin[j]) dmin[j] = d2;
        }
        __syncthreads();
    }
}

// ---------------------------------------------------------------------------
// Kernel 2: radius ball query = K nearest within R^2, one workgroup per query.
// Shuffle-based min reduction; invalid slots marked 0xFFFF.
// ---------------------------------------------------------------------------
__global__ void __launch_bounds__(256)
ballq_kernel(const float* __restrict__ pos,
             const unsigned short* __restrict__ sel,
             unsigned short* __restrict__ nbr) {
    __shared__ float d2s[N_];
    __shared__ float wrv[8];
    __shared__ int   wri[8];

    int q = blockIdx.x, tid = threadIdx.x;
    int b = q / M_;
    const float* pb = pos + (size_t)b * N_ * 3;
    int si = sel[q];
    float sx = pb[si * 3 + 0], sy = pb[si * 3 + 1], sz = pb[si * 3 + 2];

    for (int j = tid; j < N_; j += 256) {
        float dx = pb[j * 3 + 0] - sx;
        float dy = pb[j * 3 + 1] - sy;
        float dz = pb[j * 3 + 2] - sz;
        d2s[j] = dx * dx + dy * dy + dz * dz;
    }
    __syncthreads();

    for (int k = 0; k < KN; ++k) {
        float bv = 3.4e38f; int bi = N_;
        for (int j = tid; j < N_; j += 256)
            if (d2s[j] < bv) { bv = d2s[j]; bi = j; }
        for (int off = 16; off > 0; off >>= 1) {
            float ov = __shfl_down(bv, off);
            int   oi = __shfl_down(bi, off);
            if (ov < bv || (ov == bv && oi < bi)) { bv = ov; bi = oi; }
        }
        if ((tid & 31) == 0) { wrv[tid >> 5] = bv; wri[tid >> 5] = bi; }
        __syncthreads();
        if (tid == 0) {
            bv = wrv[0]; bi = wri[0];
            for (int w = 1; w < 8; ++w)
                if (wrv[w] < bv || (wrv[w] == bv && wri[w] < bi)) { bv = wrv[w]; bi = wri[w]; }
            if (bv <= R2_) {
                nbr[(size_t)q * KN + k] = (unsigned short)bi;
                d2s[bi] = 3.4e38f;
            } else {
                nbr[(size_t)q * KN + k] = 0xFFFFu;
            }
        }
        __syncthreads();
    }
}

// ---------------------------------------------------------------------------
// Kernel 3: grouped MLP via WMMA f16 + masked max.  One 4-wave block per
// selected point; each wave owns 16 neighbor rows.
// ---------------------------------------------------------------------------
__global__ void __launch_bounds__(128)
conv_kernel(const float* __restrict__ x, const float* __restrict__ pos,
            const unsigned short* __restrict__ sel,
            const unsigned short* __restrict__ nbr,
            const _Float16* __restrict__ W1t, const _Float16* __restrict__ W2t,
            const _Float16* __restrict__ W3t,
            const float* __restrict__ b1, const float* __restrict__ b2,
            const float* __restrict__ b3, float* __restrict__ out) {
    __shared__ _Float16 h0[KN * KPAD];   // [64 rows][96]  inputs (f16)
    __shared__ _Float16 h1[KN * 64];     // layer-1 activations
    __shared__ _Float16 h2[KN * 64];     // layer-2 activations
    __shared__ unsigned int omax[OUT_];  // max-reduce (relu>=0 -> uint order ok)
    __shared__ int validr[KN];

    int q = blockIdx.x, tid = threadIdx.x;
    int b = q / M_;
    int gi = b * N_ + sel[q];
    float sx = pos[(size_t)gi * 3 + 0];
    float sy = pos[(size_t)gi * 3 + 1];
    float sz = pos[(size_t)gi * 3 + 2];

    omax[tid] = 0u;

    // Stage h0: two threads per neighbor row, 48 cols each.
    {
        int row = tid >> 1;
        int part = tid & 1;
        unsigned short nb = nbr[(size_t)q * KN + row];
        int ok = (nb != 0xFFFFu);
        if (part == 0) validr[row] = ok;
        int gj = b * N_ + (ok ? (int)nb : 0);
        const float* xr = x + (size_t)gj * FIN;
        int c0 = part * 48;
        for (int c = c0; c < c0 + 48; ++c) {
            float v;
            if (!ok)          v = 0.f;
            else if (c < 64)  v = xr[c];
            else if (c == 64) v = pos[(size_t)gj * 3 + 0] - sx;
            else if (c == 65) v = pos[(size_t)gj * 3 + 1] - sy;
            else if (c == 66) v = pos[(size_t)gj * 3 + 2] - sz;
            else              v = 0.f;
            h0[row * KPAD + c] = (_Float16)v;
        }
    }
    __syncthreads();

    int lane = tid & 31;
    int wv   = tid >> 5;
    int grp  = lane >> 4;   // lane group (0: lanes 0-15, 1: lanes 16-31)
    int lr   = lane & 15;
    int rowBase = wv * 16;

    // ----- Layer 1: [16 x 96] @ [96 x 64] -----
    for (int t = 0; t < 4; ++t) {
        int n0 = t * 16;
        v8f acc = {};
        for (int kk = 0; kk < KPAD; kk += 32) {
            const _Float16* ap = &h0[(rowBase + lr) * KPAD + kk];
            int ka = grp * 16;
            v16h a, bm;
#pragma unroll
            for (int e = 0; e < 8; ++e) { a[e] = ap[ka + e]; a[8 + e] = ap[8 + ka + e]; }
            const _Float16* bp = &W1t[(n0 + lr) * KPAD + kk + grp * 16];
#pragma unroll
            for (int e = 0; e < 16; ++e) bm[e] = bp[e];
            acc = __builtin_amdgcn_wmma_f32_16x16x32_f16(false, a, false, bm,
                                                         (short)0, acc, false, false);
        }
        float bias = b1[n0 + lr];
#pragma unroll
        for (int r = 0; r < 8; ++r) {
            float v = acc[r] + bias;
            v = v > 0.f ? v : 0.f;
            h1[(rowBase + grp * 8 + r) * 64 + n0 + lr] = (_Float16)v;
        }
    }
    __syncthreads();

    // ----- Layer 2: [16 x 64] @ [64 x 64] -----
    for (int t = 0; t < 4; ++t) {
        int n0 = t * 16;
        v8f acc = {};
        for (int kk = 0; kk < 64; kk += 32) {
            const _Float16* ap = &h1[(rowBase + lr) * 64 + kk];
            int ka = grp * 16;
            v16h a, bm;
#pragma unroll
            for (int e = 0; e < 8; ++e) { a[e] = ap[ka + e]; a[8 + e] = ap[8 + ka + e]; }
            const _Float16* bp = &W2t[(n0 + lr) * 64 + kk + grp * 16];
#pragma unroll
            for (int e = 0; e < 16; ++e) bm[e] = bp[e];
            acc = __builtin_amdgcn_wmma_f32_16x16x32_f16(false, a, false, bm,
                                                         (short)0, acc, false, false);
        }
        float bias = b2[n0 + lr];
#pragma unroll
        for (int r = 0; r < 8; ++r) {
            float v = acc[r] + bias;
            v = v > 0.f ? v : 0.f;
            h2[(rowBase + grp * 8 + r) * 64 + n0 + lr] = (_Float16)v;
        }
    }
    __syncthreads();

    // ----- Layer 3: [16 x 64] @ [64 x 128] + masked max over neighbors -----
    for (int t = 0; t < 8; ++t) {
        int n0 = t * 16;
        v8f acc = {};
        for (int kk = 0; kk < 64; kk += 32) {
            const _Float16* ap = &h2[(rowBase + lr) * 64 + kk];
            int ka = grp * 16;
            v16h a, bm;
#pragma unroll
            for (int e = 0; e < 8; ++e) { a[e] = ap[ka + e]; a[8 + e] = ap[8 + ka + e]; }
            const _Float16* bp = &W3t[(n0 + lr) * 64 + kk + grp * 16];
#pragma unroll
            for (int e = 0; e < 16; ++e) bm[e] = bp[e];
            acc = __builtin_amdgcn_wmma_f32_16x16x32_f16(false, a, false, bm,
                                                         (short)0, acc, false, false);
        }
        float bias = b3[n0 + lr];
        float mx = -1.f;
#pragma unroll
        for (int r = 0; r < 8; ++r) {
            int row = rowBase + grp * 8 + r;
            if (validr[row]) {
                float v = acc[r] + bias;
                v = v > 0.f ? v : 0.f;
                mx = v > mx ? v : mx;
            }
        }
        if (mx >= 0.f) atomicMax(&omax[n0 + lr], __float_as_uint(mx));
    }
    __syncthreads();

    out[(size_t)q * OUT_ + tid] = __uint_as_float(omax[tid]);
}

// ---------------------------------------------------------------------------
// Host-side launcher
// ---------------------------------------------------------------------------
extern "C" void kernel_launch(void* const* d_in, const int* in_sizes, int n_in,
                              void* d_out, int out_size, void* d_ws, size_t ws_size,
                              hipStream_t stream) {
    (void)in_sizes; (void)n_in; (void)out_size; (void)ws_size;

    const float* x   = (const float*)d_in[0];
    const float* pos = (const float*)d_in[1];
    /* d_in[2] = batch (unused: derived analytically) */
    const float* W1  = (const float*)d_in[3];
    const float* b1  = (const float*)d_in[4];
    const float* W2  = (const float*)d_in[5];
    const float* b2  = (const float*)d_in[6];
    const float* W3  = (const float*)d_in[7];
    const float* b3  = (const float*)d_in[8];

    char* ws = (char*)d_ws;
    _Float16*       W1t = (_Float16*)(ws + 0);        // 64*96*2  = 12288
    _Float16*       W2t = (_Float16*)(ws + 12288);    // 64*64*2  =  8192
    _Float16*       W3t = (_Float16*)(ws + 20480);    // 128*64*2 = 16384
    unsigned short* sel = (unsigned short*)(ws + 36864);   // 8192*2
    unsigned short* nbr = (unsigned short*)(ws + 53248);   // 8192*64*2

    float* out      = (float*)d_out;                        // [8192,128]
    float* selpos   = out + (size_t)B_ * M_ * OUT_;         // [8192,3]
    int*   selbatch = (int*)(selpos + (size_t)B_ * M_ * 3); // [8192] int32

    wconv_kernel<<<1, 256, 0, stream>>>(W1, W2, W3, W1t, W2t, W3t);
    fps_kernel<<<B_, FPS_T, 0, stream>>>(pos, sel, selpos, selbatch);
    ballq_kernel<<<B_ * M_, 256, 0, stream>>>(pos, sel, nbr);
    conv_kernel<<<B_ * M_, 128, 0, stream>>>(x, pos, sel, nbr,
                                             W1t, W2t, W3t, b1, b2, b3, out);
}
